// MoeGate_73297911874180
// MI455X (gfx1250) — matrix operand order repaired and checked by reading, not compiled, not run on hardware
//
#include <hip/hip_runtime.h>
#include <hip/hip_bf16.h>

typedef __attribute__((ext_vector_type(16))) __bf16 bf16x16;
typedef __attribute__((ext_vector_type(8)))  float  f32x8;

#define HDIM      2048
#define NEXP      64
#define TOPK      8
#define TOK_PER_WG 128
#define SROW      65            // padded row stride (floats) to avoid bank conflicts
#define EPS_F     1e-9f
#define ALPHA_F   0.001f

__device__ __forceinline__ float sigmoidf(float v) {
    return 1.0f / (1.0f + __expf(-v));
}

__device__ __forceinline__ bf16x16 pack_a(float4 f0, float4 f1, float4 f2, float4 f3) {
    bf16x16 a;
    a[0]  = (__bf16)f0.x; a[1]  = (__bf16)f0.y; a[2]  = (__bf16)f0.z; a[3]  = (__bf16)f0.w;
    a[4]  = (__bf16)f1.x; a[5]  = (__bf16)f1.y; a[6]  = (__bf16)f1.z; a[7]  = (__bf16)f1.w;
    a[8]  = (__bf16)f2.x; a[9]  = (__bf16)f2.y; a[10] = (__bf16)f2.z; a[11] = (__bf16)f2.w;
    a[12] = (__bf16)f3.x; a[13] = (__bf16)f3.y; a[14] = (__bf16)f3.z; a[15] = (__bf16)f3.w;
    return a;
}

// ---------------------------------------------------------------------------
// Kernel 1: convert router weight [64][2048] f32 -> bf16 (row-major, same layout)
// ---------------------------------------------------------------------------
__global__ __launch_bounds__(256) void moe_prep_weight(const float* __restrict__ w,
                                                       __bf16* __restrict__ wbf,
                                                       int n) {
    int i = blockIdx.x * blockDim.x + threadIdx.x;
    if (i < n) wbf[i] = (__bf16)w[i];
}

// ---------------------------------------------------------------------------
// Kernel 2: main — bf16 WMMA GEMM (x · W^T) + sigmoid + top-8 + per-WG aux stats
// ---------------------------------------------------------------------------
__global__ __launch_bounds__(256) void moe_gate_main(const float* __restrict__ x,
                                                     const __bf16* __restrict__ wbf,
                                                     float* __restrict__ out,
                                                     float* __restrict__ counts_part,
                                                     float* __restrict__ prob_part,
                                                     int ntok) {
    __shared__ float              s_scores[TOK_PER_WG * SROW]; // sigmoid scores
    __shared__ float              s_ssum[TOK_PER_WG];          // per-token score sum
    __shared__ unsigned long long s_mask[TOK_PER_WG];          // chosen-expert bitmask
    __shared__ float              s_prob4[4][NEXP];            // prob_raw partials

    const int tid  = threadIdx.x;
    const int wave = tid >> 5;
    const int lane = tid & 31;
    const int m    = lane & 15;     // row within 16 (token) / expert column
    const int hi   = lane >> 4;     // half-wave select
    const int wg_tok = blockIdx.x * TOK_PER_WG;

    // ---------------- GEMM: 16 tokens x 64 experts per wave ----------------
    const int tok0 = wg_tok + wave * 16;
    const float* xrow = x + (size_t)(tok0 + m) * HDIM;
    const int akb = hi * 16;   // A K-offset: lanes 0-15 -> K 0..15, 16-31 -> K 16..31
    const int bkb = hi * 16;   // B K-offset: lanes 0-15 -> K 0..15, 16-31 -> K 16..31
    const __bf16* brow = wbf + (size_t)m * HDIM + bkb;

    f32x8 acc0 = {}, acc1 = {}, acc2 = {}, acc3 = {};

    // Prologue: load A tile for k0 = 0
    const float* ap = xrow + akb;
    float4 c0 = *(const float4*)(ap);
    float4 c1 = *(const float4*)(ap + 4);
    float4 c2 = *(const float4*)(ap + 8);
    float4 c3 = *(const float4*)(ap + 12);

    // Steady state: B loads first, then next-iteration A loads (so the WMMA
    // wait only drains B while 4 A-loads stay in flight), then convert + 4 WMMA.
    for (int k0 = 0; k0 < HDIM - 32; k0 += 32) {
        const __bf16* bp = brow + k0;
        bf16x16 b0 = *(const bf16x16*)(bp + (size_t)0 * 16 * HDIM);
        bf16x16 b1 = *(const bf16x16*)(bp + (size_t)1 * 16 * HDIM);
        bf16x16 b2 = *(const bf16x16*)(bp + (size_t)2 * 16 * HDIM);
        bf16x16 b3 = *(const bf16x16*)(bp + (size_t)3 * 16 * HDIM);

        __builtin_prefetch(xrow + k0 + 64 + akb, 0, 3);   // WGP-scope, 2 steps ahead

        const float* np = xrow + k0 + 32 + akb;           // double-buffered A
        float4 n0 = *(const float4*)(np);
        float4 n1 = *(const float4*)(np + 4);
        float4 n2 = *(const float4*)(np + 8);
        float4 n3 = *(const float4*)(np + 12);

        bf16x16 a = pack_a(c0, c1, c2, c3);
        acc0 = __builtin_amdgcn_wmma_f32_16x16x32_bf16(false, a, false, b0, (short)0, acc0, false, false);
        acc1 = __builtin_amdgcn_wmma_f32_16x16x32_bf16(false, a, false, b1, (short)0, acc1, false, false);
        acc2 = __builtin_amdgcn_wmma_f32_16x16x32_bf16(false, a, false, b2, (short)0, acc2, false, false);
        acc3 = __builtin_amdgcn_wmma_f32_16x16x32_bf16(false, a, false, b3, (short)0, acc3, false, false);

        c0 = n0; c1 = n1; c2 = n2; c3 = n3;
    }

    // Epilogue: last K-chunk
    {
        const __bf16* bp = brow + (HDIM - 32);
        bf16x16 b0 = *(const bf16x16*)(bp + (size_t)0 * 16 * HDIM);
        bf16x16 b1 = *(const bf16x16*)(bp + (size_t)1 * 16 * HDIM);
        bf16x16 b2 = *(const bf16x16*)(bp + (size_t)2 * 16 * HDIM);
        bf16x16 b3 = *(const bf16x16*)(bp + (size_t)3 * 16 * HDIM);

        bf16x16 a = pack_a(c0, c1, c2, c3);
        acc0 = __builtin_amdgcn_wmma_f32_16x16x32_bf16(false, a, false, b0, (short)0, acc0, false, false);
        acc1 = __builtin_amdgcn_wmma_f32_16x16x32_bf16(false, a, false, b1, (short)0, acc1, false, false);
        acc2 = __builtin_amdgcn_wmma_f32_16x16x32_bf16(false, a, false, b2, (short)0, acc2, false, false);
        acc3 = __builtin_amdgcn_wmma_f32_16x16x32_bf16(false, a, false, b3, (short)0, acc3, false, false);
    }

    // C layout: VGPR r, lanes 0-15 -> M=r, lanes 16-31 -> M=r+8, N=lane&15
    #pragma unroll
    for (int r = 0; r < 8; ++r) {
        int tl = wave * 16 + r + hi * 8;
        float* dst = &s_scores[tl * SROW];
        dst[0 * 16 + m] = sigmoidf(acc0[r]);
        dst[1 * 16 + m] = sigmoidf(acc1[r]);
        dst[2 * 16 + m] = sigmoidf(acc2[r]);
        dst[3 * 16 + m] = sigmoidf(acc3[r]);
    }
    __syncthreads();

    // ---------------- Phase A: per-token top-8, normalize, write out -------
    const size_t nflat = (size_t)ntok * TOPK;
    if (tid < TOK_PER_WG) {
        const int tl = tid;
        const int n  = wg_tok + tl;
        const float* srow = &s_scores[tl * SROW];

        float ssum = 0.0f;
        #pragma unroll
        for (int e = 0; e < NEXP; ++e) ssum += srow[e];
        s_ssum[tl] = ssum;

        unsigned long long chosen = 0ull;
        float vals[TOPK]; int ids[TOPK];
        float wsum = 0.0f;
        #pragma unroll
        for (int j = 0; j < TOPK; ++j) {
            float best = -1.0f; int bi = 0;
            for (int e = 0; e < NEXP; ++e) {
                if ((chosen >> e) & 1ull) continue;
                float v = srow[e];
                if (v > best) { best = v; bi = e; }   // strict > == lowest-index tie-break
            }
            chosen |= (1ull << bi);
            vals[j] = best; ids[j] = bi; wsum += best;
        }
        s_mask[tl] = chosen;

        float inv = 1.0f / (wsum + EPS_F);
        #pragma unroll
        for (int j = 0; j < TOPK; ++j) {
            out[(size_t)n * TOPK + j]         = (float)ids[j];     // topk_idx
            out[nflat + (size_t)n * TOPK + j] = vals[j] * inv;     // topk_weight
        }
    }
    __syncthreads();

    // ---------------- Phase B: prob_raw partial sums (deterministic) -------
    {
        const int e = tid & (NEXP - 1);
        const int g = tid >> 6;            // 0..3, 32 tokens each
        float partial = 0.0f;
        for (int mm = g * 32; mm < g * 32 + 32; ++mm)
            partial += s_scores[mm * SROW + e] / (s_ssum[mm] + EPS_F);
        s_prob4[g][e] = partial;
    }
    __syncthreads();

    // ---------------- Per-WG stats: counts via bitmask, prob partials ------
    if (tid < NEXP) {
        float c = 0.0f;
        for (int mm = 0; mm < TOK_PER_WG; ++mm)
            c += (float)((s_mask[mm] >> tid) & 1ull);
        float p = s_prob4[0][tid] + s_prob4[1][tid] + s_prob4[2][tid] + s_prob4[3][tid];
        counts_part[(size_t)blockIdx.x * NEXP + tid] = c;
        prob_part[(size_t)blockIdx.x * NEXP + tid]   = p;
    }
}

// ---------------------------------------------------------------------------
// Kernel 3: deterministic final reduction -> aux_loss scalar
// ---------------------------------------------------------------------------
__global__ __launch_bounds__(64) void moe_finalize(const float* __restrict__ counts_part,
                                                   const float* __restrict__ prob_part,
                                                   float* __restrict__ aux_out,
                                                   int ntok, int nwg) {
    __shared__ float s_c[NEXP];
    __shared__ float s_p[NEXP];
    int t = threadIdx.x;
    if (t < NEXP) {
        float c = 0.0f, p = 0.0f;
        for (int w = 0; w < nwg; ++w) {
            c += counts_part[(size_t)w * NEXP + t];
            p += prob_part[(size_t)w * NEXP + t];
        }
        s_c[t] = c; s_p[t] = p;
    }
    __syncthreads();
    if (t == 0) {
        float invn = 1.0f / (float)ntok;
        float psum = 0.0f;
        for (int e = 0; e < NEXP; ++e) psum += s_p[e] * invn;   // prob.sum()
        float aux = 0.0f;
        for (int e = 0; e < NEXP; ++e) {
            float load = s_c[e] / ((float)ntok * (float)TOPK);
            float prob = (s_p[e] * invn) / (psum + EPS_F);
            aux += load * prob;
        }
        aux_out[0] = aux * ALPHA_F * (float)NEXP;
    }
}

// ---------------------------------------------------------------------------
extern "C" void kernel_launch(void* const* d_in, const int* in_sizes, int n_in,
                              void* d_out, int out_size, void* d_ws, size_t ws_size,
                              hipStream_t stream) {
    const float* x = (const float*)d_in[0];   // [B*S, 2048] f32
    const float* w = (const float*)d_in[1];   // [64, 2048] f32
    float* out = (float*)d_out;               // [N*8 idx | N*8 weight | aux]

    const int ntok = in_sizes[0] / HDIM;      // 16384
    const int nwg  = ntok / TOK_PER_WG;       // 128

    // Workspace layout:
    //   [0, 256KB)          : bf16 weight [64][2048]
    //   [256KB, +32KB)      : per-WG expert counts  [nwg][64] f32
    //   [256KB+32KB, +32KB) : per-WG prob partials  [nwg][64] f32
    __bf16* wbf = (__bf16*)d_ws;
    float* counts_part = (float*)((char*)d_ws + (size_t)NEXP * HDIM * sizeof(__bf16));
    float* prob_part   = counts_part + (size_t)nwg * NEXP;

    const int wn = NEXP * HDIM;
    moe_prep_weight<<<(wn + 255) / 256, 256, 0, stream>>>(w, wbf, wn);

    moe_gate_main<<<nwg, 256, 0, stream>>>(x, wbf, out, counts_part, prob_part, ntok);

    moe_finalize<<<1, 64, 0, stream>>>(counts_part, prob_part,
                                       out + 2 * (size_t)ntok * TOPK, ntok, nwg);
}
